// MultiHeadAttention_2542620639331
// MI455X (gfx1250) — matrix operand order repaired and checked by reading, not compiled, not run on hardware
//
#include <hip/hip_runtime.h>

// ---------------------------------------------------------------------------
// MHA forward for MI455X (gfx1250, wave32, WMMA).
//  - bf16 operands / f32 accumulate on v_wmma_f32_16x16x32_bf16
//  - flash attention computes S^T = K*Q^T so softmax is per-lane + shfl_xor(16)
//  - K/V tiles staged block-wide into LDS via global_load_async_to_lds_b128
//  - GEMM epilogue templated on output mode (no runtime branching)
// ---------------------------------------------------------------------------

typedef __bf16 bf16_t;
typedef __attribute__((ext_vector_type(16))) __bf16 bf16x16;
typedef __attribute__((ext_vector_type(8)))  __bf16 bf16x8;
typedef __attribute__((ext_vector_type(4)))  __bf16 bf16x4;
typedef __attribute__((ext_vector_type(8)))  float  f32x8;
typedef __attribute__((ext_vector_type(4)))  int    i32x4;   // native vector for builtins

#define D_MODEL   1024
#define NUM_HEADS 16
#define HEAD_DIM  64
#define BATCH     4
#define SEQ       2048
#define MTOT      (BATCH * SEQ)        // 8192 rows

#if __has_builtin(__builtin_amdgcn_global_load_async_to_lds_b128)
#define USE_ASYNC_COPY 1
#else
#define USE_ASYNC_COPY 0
#endif

__device__ __forceinline__ f32x8 zero8() {
  f32x8 z;
#pragma unroll
  for (int i = 0; i < 8; ++i) z[i] = 0.0f;
  return z;
}

__device__ __forceinline__ f32x8 wmma_bf16(bf16x16 a, bf16x16 b, f32x8 c) {
  // D = A(16x32 bf16) * B(32x16 bf16) + C(16x16 f32)
  return __builtin_amdgcn_wmma_f32_16x16x32_bf16(
      false, a, false, b, (short)0, c, false, false);
}

union frag_u { bf16x16 v; bf16x8 h[2]; };

// A operand (M x K, 16x32): lane l holds row m = l&15.
// Per ISA table: lanes 0-15 get K={0..7,16..23}, lanes 16-31 get K={8..15,24..31}.
__device__ __forceinline__ bf16x16 load_frag_a(const bf16_t* base, int ld) {
  const int l  = threadIdx.x & 31;
  const int m  = l & 15;
  const int hi = l >> 4;
  const bf16_t* p = base + (size_t)m * ld + hi * 8;
  frag_u u;
  u.h[0] = *(const bf16x8*)(p);
  u.h[1] = *(const bf16x8*)(p + 16);
  return u.v;
}

// B operand (K x N = 32x16). Source stored as N x K row-major (i.e. B^T).
// Lane l holds column n = l&15 with K={hi*8..hi*8+7, 16+hi*8..16+hi*8+7} (hi=l>>4).
__device__ __forceinline__ bf16x16 load_frag_b(const bf16_t* base, int ld) {
  const int l  = threadIdx.x & 31;
  const int n  = l & 15;
  const int hi = l >> 4;
  const bf16_t* p = base + (size_t)n * ld + hi * 8;
  frag_u u;
  u.h[0] = *(const bf16x8*)(p);
  u.h[1] = *(const bf16x8*)(p + 16);
  return u.v;
}

// 16-byte global -> LDS stage (gfx1250 async-to-LDS path when available).
// Toolchain declares the builtin with native int-vector (vector_size 16) ptrs.
__device__ __forceinline__ void stage16(const bf16_t* g, bf16_t* l) {
#if USE_ASYNC_COPY
  __builtin_amdgcn_global_load_async_to_lds_b128(
      (i32x4*)(void*)(g), (i32x4*)(void*)(l), 0, 0);
#else
  *(i32x4*)(void*)l = *(const i32x4*)(const void*)g;
#endif
}

__device__ __forceinline__ void stage_fence() {
#if USE_ASYNC_COPY
  asm volatile("s_wait_asynccnt 0" ::: "memory");
#endif
}

// ---------------------------------------------------------------------------
// fp32 -> bf16 conversion (vectorized x4)
// ---------------------------------------------------------------------------
__global__ __launch_bounds__(256) void cvt_f32_bf16(
    const float* __restrict__ in, bf16_t* __restrict__ out, int n4) {
  int i = blockIdx.x * 256 + threadIdx.x;
  if (i < n4) {
    float4 v = ((const float4*)in)[i];
    bf16x4 o;
    o[0] = (bf16_t)v.x; o[1] = (bf16_t)v.y;
    o[2] = (bf16_t)v.z; o[3] = (bf16_t)v.w;
    ((bf16x4*)out)[i] = o;
  }
}

// ---------------------------------------------------------------------------
// GEMM: out = A[8192x1024] * B^T (B stored [1024 x 1024] N-major) + bias
// MODE 0: bf16 out, scatter to [B,H,S,Dh]
// MODE 1: bf16 out, scatter to [B,H,Dh,S]   (transposed V)
// MODE 2: fp32 out, plain row-major [M,N]
// Block = 256 thr (8 waves) -> 128x128 tile; wave = 32x64 (2x4 WMMA tiles).
// ---------------------------------------------------------------------------
template <int MODE>
__global__ __launch_bounds__(256) void gemm_wmma_bf16(
    const bf16_t* __restrict__ A, const bf16_t* __restrict__ B,
    const float* __restrict__ bias, void* __restrict__ out) {
  const int lane = threadIdx.x & 31;
  const int w    = threadIdx.x >> 5;
  const int wm   = w >> 1;          // 0..3
  const int wn   = w & 1;           // 0..1
  const int hi   = lane >> 4;
  const int mbase = blockIdx.y * 128 + wm * 32;
  const int nbase = blockIdx.x * 128 + wn * 64;

  f32x8 acc[2][4];
#pragma unroll
  for (int i = 0; i < 2; ++i)
#pragma unroll
    for (int j = 0; j < 4; ++j) acc[i][j] = zero8();

  const bf16_t* Arow = A + (size_t)mbase * D_MODEL;
  const bf16_t* Brow = B + (size_t)nbase * D_MODEL;

  for (int k0 = 0; k0 < D_MODEL; k0 += 32) {
    bf16x16 af[2], bfr[4];
#pragma unroll
    for (int mt = 0; mt < 2; ++mt)
      af[mt] = load_frag_a(Arow + (size_t)(mt * 16) * D_MODEL + k0, D_MODEL);
#pragma unroll
    for (int nt = 0; nt < 4; ++nt)
      bfr[nt] = load_frag_b(Brow + (size_t)(nt * 16) * D_MODEL + k0, D_MODEL);
#pragma unroll
    for (int mt = 0; mt < 2; ++mt)
#pragma unroll
      for (int nt = 0; nt < 4; ++nt)
        acc[mt][nt] = wmma_bf16(af[mt], bfr[nt], acc[mt][nt]);
  }

#pragma unroll
  for (int mt = 0; mt < 2; ++mt) {
#pragma unroll
    for (int nt = 0; nt < 4; ++nt) {
      const int n  = nbase + nt * 16 + (lane & 15);
      const float bv = bias[n];
      const int h = n >> 6, d = n & 63;
#pragma unroll
      for (int r = 0; r < 8; ++r) {
        const int m = mbase + mt * 16 + hi * 8 + r;
        const float val = acc[mt][nt][r] + bv;
        const int b = m >> 11, s = m & 2047;
        if (MODE == 0) {
          ((bf16_t*)out)[(((size_t)(b * NUM_HEADS + h) * SEQ + s) * HEAD_DIM) + d] =
              (bf16_t)val;
        } else if (MODE == 1) {
          ((bf16_t*)out)[(((size_t)(b * NUM_HEADS + h) * HEAD_DIM + d) * SEQ) + s] =
              (bf16_t)val;
        } else {
          ((float*)out)[(size_t)m * D_MODEL + n] = val;
        }
      }
    }
  }
}

// ---------------------------------------------------------------------------
// Flash attention. Block = 8 waves; each wave owns 16 queries; the whole
// block streams the SAME 32-key block, so K / V^T tiles are staged once per
// iteration into LDS (4 KB each, one b128 per thread) and fragments are read
// back with ds_load_b128.
//   S^T tile  = K_blk(16x64) x Q^T(64x16)          (2 M-tiles x 2 K-steps)
//   ctx^T     = V^T(16x32)   x P^T(32x16)          (4 dh tiles)
// q,k: [B,H,S,Dh] bf16; vT: [B,H,Dh,S] bf16; ctx out: [B,S,H*Dh] bf16.
// ---------------------------------------------------------------------------
__global__ __launch_bounds__(256) void attn_flash_wmma(
    const bf16_t* __restrict__ q, const bf16_t* __restrict__ k,
    const bf16_t* __restrict__ vT, bf16_t* __restrict__ ctx) {
  __shared__ bf16_t ldsK[32 * 64];        // K rows [key][dh]      4 KB
  __shared__ bf16_t ldsV[64 * 32];        // V^T rows [dh][key]    4 KB
  __shared__ bf16_t Plds[8 * 16 * 32];    // per-wave P            8 KB

  const int tid  = threadIdx.x;
  const int lane = tid & 31;
  const int w    = tid >> 5;
  const int n    = lane & 15;              // query index inside tile
  const int hi   = lane >> 4;
  const int bh   = blockIdx.y;
  const int b    = bh >> 4, h = bh & 15;
  const int q0   = blockIdx.x * 128 + w * 16;

  const bf16_t* qh = q  + (size_t)bh * SEQ * HEAD_DIM;
  const bf16_t* kh = k  + (size_t)bh * SEQ * HEAD_DIM;
  const bf16_t* vh = vT + (size_t)bh * HEAD_DIM * SEQ;
  bf16_t* plw = Plds + w * (16 * 32);

  // staging assignment: one 16B chunk per thread per tile
  const int krow = tid >> 3, kcol = (tid & 7) * 8;   // 32 rows x 64 el
  const int vrow = tid >> 2, vcol = (tid & 3) * 8;   // 64 rows x 32 el

  // Q^T operand fragments, resident for the whole key loop.
  bf16x16 qf[2];
  qf[0] = load_frag_b(qh + (size_t)q0 * HEAD_DIM +  0, HEAD_DIM);
  qf[1] = load_frag_b(qh + (size_t)q0 * HEAD_DIM + 32, HEAD_DIM);

  float m_i = -1e30f, l_i = 0.0f;
  f32x8 accO[4];
#pragma unroll
  for (int t = 0; t < 4; ++t) accO[t] = zero8();

  const float scale = 0.125f;              // 1/sqrt(64)

  for (int kb0 = 0; kb0 < SEQ; kb0 += 32) {
    // ---- stage K / V^T tiles for the whole block ----
    __syncthreads();                       // previous iter readers done
    stage16(kh + (size_t)(kb0 + krow) * HEAD_DIM + kcol, ldsK + krow * 64 + kcol);
    stage16(vh + (size_t)vrow * SEQ + kb0 + vcol,        ldsV + vrow * 32 + vcol);
    stage_fence();
    __syncthreads();

    // ---- scores S^T (keys x queries), two 16-key sub-tiles ----
    f32x8 s0 = zero8(), s1 = zero8();
    {
      bf16x16 ka;
      ka = load_frag_a(ldsK +  0 * 64 +  0, 64); s0 = wmma_bf16(ka, qf[0], s0);
      ka = load_frag_a(ldsK +  0 * 64 + 32, 64); s0 = wmma_bf16(ka, qf[1], s0);
      ka = load_frag_a(ldsK + 16 * 64 +  0, 64); s1 = wmma_bf16(ka, qf[0], s1);
      ka = load_frag_a(ldsK + 16 * 64 + 32, 64); s1 = wmma_bf16(ka, qf[1], s1);
    }

    // ---- online softmax (per lane == per query, partner lane = lane^16) ----
    float bm = -1e30f;
#pragma unroll
    for (int e = 0; e < 8; ++e) {
      s0[e] *= scale; s1[e] *= scale;
      bm = fmaxf(bm, fmaxf(s0[e], s1[e]));
    }
    bm = fmaxf(bm, __shfl_xor(bm, 16));
    const float m_new = fmaxf(m_i, bm);
    const float alpha = __expf(m_i - m_new);

    float p0[8], p1[8], rsum = 0.0f;
#pragma unroll
    for (int e = 0; e < 8; ++e) {
      p0[e] = __expf(s0[e] - m_new);
      p1[e] = __expf(s1[e] - m_new);
      rsum += p0[e] + p1[e];
    }
    rsum += __shfl_xor(rsum, 16);
    l_i = l_i * alpha + rsum;
    m_i = m_new;

#pragma unroll
    for (int t = 0; t < 4; ++t)
#pragma unroll
      for (int e = 0; e < 8; ++e) accO[t][e] *= alpha;

    // ---- P (bf16) -> wave-private LDS as [query n][key 0..31] ----
    bf16x8 pk0, pk1;
#pragma unroll
    for (int e = 0; e < 8; ++e) { pk0[e] = (bf16_t)p0[e]; pk1[e] = (bf16_t)p1[e]; }
    *(bf16x8*)(plw + n * 32 + hi * 8)      = pk0;   // keys hi*8..hi*8+7
    *(bf16x8*)(plw + n * 32 + 16 + hi * 8) = pk1;   // keys 16+hi*8..

    frag_u pu;
    pu.h[0] = *(const bf16x8*)(plw + n * 32 + hi * 8);
    pu.h[1] = *(const bf16x8*)(plw + n * 32 + hi * 8 + 16);
    const bf16x16 pf = pu.v;

    // ---- ctx^T += V^T * P^T over 4 dh tiles (V^T from LDS) ----
#pragma unroll
    for (int t = 0; t < 4; ++t) {
      bf16x16 vf = load_frag_a(ldsV + t * 16 * 32, 32);
      accO[t] = wmma_bf16(vf, pf, accO[t]);
    }
  }

  // ---- normalize and write ctx[b][q][h*64 + dh] (packed 16B stores) ----
  const float inv = 1.0f / l_i;
#pragma unroll
  for (int t = 0; t < 4; ++t) {
    bf16x8 o;
#pragma unroll
    for (int e = 0; e < 8; ++e) o[e] = (bf16_t)(accO[t][e] * inv);
    *(bf16x8*)(ctx + (size_t)(b * SEQ + q0 + n) * D_MODEL +
               h * HEAD_DIM + t * 16 + hi * 8) = o;
  }
}

// ---------------------------------------------------------------------------
// Host-side launcher
// ---------------------------------------------------------------------------
extern "C" void kernel_launch(void* const* d_in, const int* in_sizes, int n_in,
                              void* d_out, int out_size, void* d_ws, size_t ws_size,
                              hipStream_t stream) {
  const float* x  = (const float*)d_in[0];
  const float* Wq = (const float*)d_in[1];
  const float* bq = (const float*)d_in[2];
  const float* Wk = (const float*)d_in[3];
  const float* bk = (const float*)d_in[4];
  const float* Wv = (const float*)d_in[5];
  const float* bv = (const float*)d_in[6];
  const float* Wo = (const float*)d_in[7];
  const float* bo = (const float*)d_in[8];

  const size_t XN = (size_t)MTOT * D_MODEL;    // 8,388,608
  const size_t WN = (size_t)D_MODEL * D_MODEL; // 1,048,576

  // workspace layout (bf16 elements); total ~92 MB
  bf16_t* ws  = (bf16_t*)d_ws;
  bf16_t* xb  = ws;
  bf16_t* Wqb = xb  + XN;
  bf16_t* Wkb = Wqb + WN;
  bf16_t* Wvb = Wkb + WN;
  bf16_t* Wob = Wvb + WN;
  bf16_t* qb  = Wob + WN;   // [B,H,S,Dh]
  bf16_t* kb  = qb  + XN;   // [B,H,S,Dh]
  bf16_t* vTb = kb  + XN;   // [B,H,Dh,S]
  bf16_t* cxb = vTb + XN;   // [B,S,H*Dh]

  // 1) fp32 -> bf16
  cvt_f32_bf16<<<dim3((unsigned)(XN / 4 / 256)), 256, 0, stream>>>(x, xb, (int)(XN / 4));
  cvt_f32_bf16<<<dim3((unsigned)(WN / 4 / 256)), 256, 0, stream>>>(Wq, Wqb, (int)(WN / 4));
  cvt_f32_bf16<<<dim3((unsigned)(WN / 4 / 256)), 256, 0, stream>>>(Wk, Wkb, (int)(WN / 4));
  cvt_f32_bf16<<<dim3((unsigned)(WN / 4 / 256)), 256, 0, stream>>>(Wv, Wvb, (int)(WN / 4));
  cvt_f32_bf16<<<dim3((unsigned)(WN / 4 / 256)), 256, 0, stream>>>(Wo, Wob, (int)(WN / 4));

  // 2) projections (q,k normal head layout; v transposed)
  dim3 gg(D_MODEL / 128, MTOT / 128);   // (8, 64)
  gemm_wmma_bf16<0><<<gg, 256, 0, stream>>>(xb, Wqb, bq, (void*)qb);
  gemm_wmma_bf16<0><<<gg, 256, 0, stream>>>(xb, Wkb, bk, (void*)kb);
  gemm_wmma_bf16<1><<<gg, 256, 0, stream>>>(xb, Wvb, bv, (void*)vTb);

  // 3) flash attention
  dim3 ga(SEQ / 128, BATCH * NUM_HEADS);  // (16, 64)
  attn_flash_wmma<<<ga, 256, 0, stream>>>(qb, kb, vTb, cxb);

  // 4) output projection -> fp32 d_out
  gemm_wmma_bf16<2><<<gg, 256, 0, stream>>>(cxb, Wob, bo, d_out);
}
